// Attention_8675833938565
// MI455X (gfx1250) — matrix operand order repaired and checked by reading, not compiled, not run on hardware
//
#include <hip/hip_runtime.h>
#include <stdint.h>
#include <stddef.h>

// ---------------------------------------------------------------------------
// Types for CDNA5 WMMA (gfx1250, wave32)
// ---------------------------------------------------------------------------
typedef __attribute__((ext_vector_type(4)))  __bf16 v4bf;
typedef __attribute__((ext_vector_type(8)))  __bf16 v8bf;
typedef __attribute__((ext_vector_type(16))) __bf16 v16bf;
typedef __attribute__((ext_vector_type(8)))  float  v8f;
typedef __attribute__((ext_vector_type(4)))  int    v4i;

#define B_   8
#define Q_   1024
#define K_   4096
#define H_   512
#define QT   32            // Q rows per workgroup (kernel 1 / kernel 3)
#define KC   64            // K chunk width (kernel 1)
#define NCHUNK (K_ / KC)   // 64

// LDS layout for kernel 1 (bytes)
#define SM_Q     0              // [QT][H_]  bf16  32768
#define SM_CROW  32768          // [KC][H_]  bf16  65536
#define SM_CCOL  98304          // [H_][KC]  bf16  65536 (transposed copy)
#define SM_S     163840         // [QT][KC]  f32    8192
#define SM_P     172032         // [QT][KC]  bf16   4096
#define SM_RED   176128         // [QT][8]   f32    1024
#define SM_M     177152         // [QT]      f32     128
#define SM_L     177280         // [QT]      f32     128
#define SM_A     177408         // [QT]      f32     128
#define SMEM_SZ  177536

static __device__ __forceinline__ v16bf cat16(v8bf lo, v8bf hi) {
  return __builtin_shufflevector(lo, hi, 0,1,2,3,4,5,6,7,8,9,10,11,12,13,14,15);
}
static __device__ __forceinline__ v16bf ld16(const __bf16* p0, const __bf16* p1) {
  v8bf lo = *(const v8bf*)p0;
  v8bf hi = *(const v8bf*)p1;
  return cat16(lo, hi);
}
static __device__ __forceinline__ v8f wmma_bf16(v16bf a, v16bf b, v8f c) {
  return __builtin_amdgcn_wmma_f32_16x16x32_bf16(false, a, false, b, (short)0, c,
                                                 false, false);
}

// ---------------------------------------------------------------------------
// Async global -> LDS staging (CDNA5 ASYNCcnt path) with VGPR-bounce fallback.
// Builtin prototype (from compiler diagnostic): param 1 is a pointer to a
// 4 x i32 vector in the global (AS1) address space; LDS destination is AS3.
// Generic LDS pointers carry the LDS byte offset in their low 32 bits
// (aperture in [63:32]), so a 32-bit inttoptr yields a valid as(3) pointer.
// ---------------------------------------------------------------------------
#if __has_builtin(__builtin_amdgcn_global_load_async_to_lds_b128)
#define HAVE_ASYNC_LDS 1
typedef __attribute__((address_space(1))) v4i g_v4i;
typedef __attribute__((address_space(3))) v4i l_v4i;
static __device__ __forceinline__ void copy16_to_lds(const __bf16* g, __bf16* l) {
  __builtin_amdgcn_global_load_async_to_lds_b128(
      (g_v4i*)(uintptr_t)g, (l_v4i*)(uint32_t)(uintptr_t)l, 0, 0);
}
static __device__ __forceinline__ void wait_lds_copies() {
#if __has_builtin(__builtin_amdgcn_s_wait_asynccnt)
  __builtin_amdgcn_s_wait_asynccnt(0);
#else
  asm volatile("s_wait_asynccnt 0x0" ::: "memory");
#endif
}
#else
static __device__ __forceinline__ void copy16_to_lds(const __bf16* g, __bf16* l) {
  *(v8bf*)l = *(const v8bf*)g;
}
static __device__ __forceinline__ void wait_lds_copies() {}
#endif

// ---------------------------------------------------------------------------
// Kernel 0a: f32 -> bf16 conversion (vectorized x4)
// ---------------------------------------------------------------------------
__global__ void __launch_bounds__(256)
cvt_bf16_kernel(const float* __restrict__ src, __bf16* __restrict__ dst, int n4) {
  int i = blockIdx.x * 256 + threadIdx.x;
  if (i < n4) {
    float4 v = ((const float4*)src)[i];
    v4bf o;
    o[0] = (__bf16)v.x; o[1] = (__bf16)v.y; o[2] = (__bf16)v.z; o[3] = (__bf16)v.w;
    ((v4bf*)dst)[i] = o;
  }
}

// ---------------------------------------------------------------------------
// Kernel 0b: context transpose [B,K,H] -> [B,H,K] (bf16, 32x32 LDS tiles)
// Done once so kernel 1 can stage both LDS operand layouts as pure
// contiguous copies (enables async-to-LDS, removes all scalar DS stores).
// ---------------------------------------------------------------------------
__global__ void __launch_bounds__(256)
transpose_ctx_kernel(const __bf16* __restrict__ src, __bf16* __restrict__ dst) {
  __shared__ __bf16 tile[32][33];
  const int b  = blockIdx.z;
  const int k0 = blockIdx.x * 32;
  const int h0 = blockIdx.y * 32;
  const int r  = threadIdx.x >> 3;
  const int c4 = (threadIdx.x & 7) * 4;
  v4bf v = *(const v4bf*)(src + ((size_t)(b * K_ + k0 + r)) * H_ + h0 + c4);
  tile[r][c4 + 0] = v[0]; tile[r][c4 + 1] = v[1];
  tile[r][c4 + 2] = v[2]; tile[r][c4 + 3] = v[3];
  __syncthreads();
  v4bf o;
  o[0] = tile[c4 + 0][r]; o[1] = tile[c4 + 1][r];
  o[2] = tile[c4 + 2][r]; o[3] = tile[c4 + 3][r];
  *(v4bf*)(dst + ((size_t)(b * H_ + h0 + r)) * K_ + k0 + c4) = o;
}

// ---------------------------------------------------------------------------
// Kernel 1: fused scores + online softmax + mix accumulation
//   grid = B_ * (Q_/QT) = 256 blocks, block = 256 threads (8 wave32)
//   wave w: wm = w/4 selects M-half (16 rows), wn = w%4 selects
//   16-col S tile (GEMM1) / 128-col H range (GEMM2)
// ---------------------------------------------------------------------------
__global__ void __launch_bounds__(256)
attn_softmax_mix_kernel(const __bf16* __restrict__ qbf,   // [B,Q,H] bf16
                        const __bf16* __restrict__ ctx,   // [B,K,H] bf16
                        const __bf16* __restrict__ ctxT,  // [B,H,K] bf16
                        const uint8_t* __restrict__ mask, // [B,Q,K] bool
                        float* __restrict__ attn_raw,     // [B,Q,K] raw masked scores
                        __bf16* __restrict__ mixbf,       // [B,Q,H] bf16
                        float* __restrict__ ws_m,         // [B*Q] row max
                        float* __restrict__ ws_linv)      // [B*Q] 1/sumexp
{
  extern __shared__ char smem[];
  __bf16* Qs   = (__bf16*)(smem + SM_Q);
  __bf16* Crow = (__bf16*)(smem + SM_CROW);
  __bf16* Ccol = (__bf16*)(smem + SM_CCOL);
  float*  Ss   = (float*)(smem + SM_S);
  __bf16* Ps   = (__bf16*)(smem + SM_P);
  float*  red  = (float*)(smem + SM_RED);
  float*  mrow = (float*)(smem + SM_M);
  float*  lrow = (float*)(smem + SM_L);
  float*  arow = (float*)(smem + SM_A);

  const int tid   = threadIdx.x;
  const int lane  = tid & 31;
  const int w     = tid >> 5;
  const int wm    = w >> 2;      // 0..1  (M half)
  const int wn    = w & 3;       // 0..3  (N quarter)
  const int lhalf = lane >> 4;   // 0..1
  const int l15   = lane & 15;
  const int b     = blockIdx.x >> 5;          // 32 q-tiles per batch
  const int qb    = (blockIdx.x & 31) * QT;
  const int row   = tid >> 3;    // 0..31 (softmax row owner)
  const int seg   = tid & 7;     // 0..7  (8 cols each)
  const float NEGINF = -__builtin_inff();

  // --- stage Q tile to LDS (bf16) ---
  for (int it = 0; it < (QT * H_) / (256 * 8); ++it) {   // 8 iters
    int flat = (it * 256 + tid) * 8;
    int q = flat >> 9, h = flat & 511;
    *(v8bf*)(Qs + q * H_ + h) =
        *(const v8bf*)(qbf + ((size_t)(b * Q_ + qb + q)) * H_ + h);
  }
  if (tid < QT) { mrow[tid] = NEGINF; lrow[tid] = 0.f; }

  v8f macc[8];
  #pragma unroll
  for (int i = 0; i < 8; ++i) { v8f z = {}; macc[i] = z; }

  __syncthreads();

  for (int ch = 0; ch < NCHUNK; ++ch) {
    const int kb = ch * KC;

    // --- stage context chunk: both layouts are contiguous streaming copies ---
    #pragma unroll 4
    for (int it = 0; it < (KC * H_) / (256 * 8); ++it) {  // 16 iters
      int flat = (it * 256 + tid) * 8;
      int k = flat >> 9, h = flat & 511;
      const __bf16* gp = ctx + ((size_t)(b * K_ + kb + k)) * H_ + h;
      if (ch + 1 < NCHUNK)
        __builtin_prefetch(gp + (size_t)KC * H_, 0, 0);   // global_prefetch next chunk
      copy16_to_lds(gp, Crow + k * H_ + h);
    }
    #pragma unroll 4
    for (int it = 0; it < (H_ * KC) / (256 * 8); ++it) {  // 16 iters
      int flat = (it * 256 + tid) * 8;
      int h = flat >> 6, k = flat & 63;
      copy16_to_lds(ctxT + ((size_t)(b * H_ + h)) * K_ + kb + k,
                    Ccol + h * KC + k);
    }
    wait_lds_copies();
    __syncthreads();

    // --- GEMM1: each wave computes one 16x16 S tile, K-dim = H (16 steps of 32)
    v8f s = {};
    {
      const __bf16* ap = Qs   + (wm * 16 + l15) * H_;   // A row (M = l15)
      const __bf16* bp = Crow + (wn * 16 + l15) * H_;   // B col (N = l15)
      #pragma unroll
      for (int t = 0; t < 16; ++t) {
        v16bf a = ld16(ap + t * 32 + lhalf * 16, ap + t * 32 + lhalf * 16 + 8);
        v16bf bm = *(const v16bf*)(bp + t * 32 + lhalf * 16);
        s = wmma_bf16(a, bm, s);
      }
    }
    #pragma unroll
    for (int d = 0; d < 8; ++d)   // D layout: M = 8*lhalf + d, N = l15
      Ss[(wm * 16 + lhalf * 8 + d) * KC + wn * 16 + l15] = s[d];
    __syncthreads();

    // --- mask + online softmax bookkeeping (8 threads per row) ---
    float sv[8];
    {
      uint64_t mk = *(const uint64_t*)(mask +
          ((size_t)(b * Q_ + qb + row)) * K_ + kb + seg * 8);
      float lm = NEGINF;
      #pragma unroll
      for (int i = 0; i < 8; ++i) {
        float x = Ss[row * KC + seg * 8 + i];
        if ((mk >> (8 * i)) & 0xffull) x = NEGINF;
        sv[i] = x;
        lm = fmaxf(lm, x);
      }
      // raw masked scores -> attn buffer (normalized later by kernel 2)
      float4* apn = (float4*)(attn_raw +
          ((size_t)(b * Q_ + qb + row)) * K_ + kb + seg * 8);
      apn[0] = make_float4(sv[0], sv[1], sv[2], sv[3]);
      apn[1] = make_float4(sv[4], sv[5], sv[6], sv[7]);
      red[row * 8 + seg] = lm;
    }
    __syncthreads();
    if (seg == 0) {
      float cm = red[row * 8];
      #pragma unroll
      for (int i = 1; i < 8; ++i) cm = fmaxf(cm, red[row * 8 + i]);
      float mo = mrow[row];
      float mn = fmaxf(mo, cm);
      float al = (mo > NEGINF) ? expf(mo - mn) : 0.f;
      mrow[row] = mn; arow[row] = al; lrow[row] *= al;
    }
    __syncthreads();
    {
      float mn = mrow[row];
      float ps = 0.f;
      #pragma unroll
      for (int i = 0; i < 8; ++i) {
        float p = expf(sv[i] - mn);
        ps += p;
        Ps[row * KC + seg * 8 + i] = (__bf16)p;
      }
      red[row * 8 + seg] = ps;
    }
    __syncthreads();
    if (seg == 0) {
      float sum = 0.f;
      #pragma unroll
      for (int i = 0; i < 8; ++i) sum += red[row * 8 + i];
      lrow[row] += sum;
    }
    __syncthreads();

    // --- rescale mix accumulators by alpha(row) ---
    {
      float al[8];
      #pragma unroll
      for (int d = 0; d < 8; ++d) al[d] = arow[wm * 16 + lhalf * 8 + d];
      #pragma unroll
      for (int nt = 0; nt < 8; ++nt)
        #pragma unroll
        for (int d = 0; d < 8; ++d) macc[nt][d] *= al[d];
    }

    // --- GEMM2: mix += P(16xKC) @ C(KCx512); wave covers h in [wn*128, +128)
    {
      const __bf16* pr = Ps + (wm * 16 + l15) * KC;
      #pragma unroll
      for (int ks = 0; ks < 2; ++ks) {
        v16bf a = ld16(pr + ks * 32 + lhalf * 16, pr + ks * 32 + lhalf * 16 + 8);
        #pragma unroll
        for (int nt = 0; nt < 8; ++nt) {
          int h = wn * 128 + nt * 16 + l15;
          v16bf bm = *(const v16bf*)(Ccol + h * KC + ks * 32 + lhalf * 16);
          macc[nt] = wmma_bf16(a, bm, macc[nt]);
        }
      }
    }
    __syncthreads();
  }

  // --- epilogue: normalize mix, emit bf16 mix + (m, 1/l) ---
  if (tid < QT) {
    float li = 1.0f / lrow[tid];
    ws_m[b * Q_ + qb + tid]    = mrow[tid];
    ws_linv[b * Q_ + qb + tid] = li;
    lrow[tid] = li;   // reuse as 1/l
  }
  __syncthreads();
  #pragma unroll
  for (int nt = 0; nt < 8; ++nt) {
    int h = wn * 128 + nt * 16 + l15;
    #pragma unroll
    for (int d = 0; d < 8; ++d) {
      int m = wm * 16 + lhalf * 8 + d;
      float v = macc[nt][d] * lrow[m];
      mixbf[((size_t)(b * Q_ + qb + m)) * H_ + h] = (__bf16)v;
    }
  }
}

// ---------------------------------------------------------------------------
// Kernel 2: attn = exp(S - m) * (1/l), in place over raw scores (L2-hot)
// ---------------------------------------------------------------------------
__global__ void __launch_bounds__(256)
softmax_norm_kernel(float* __restrict__ attn,
                    const float* __restrict__ ws_m,
                    const float* __restrict__ ws_linv) {
  size_t i4 = (size_t)blockIdx.x * 256 + threadIdx.x;
  size_t base = i4 * 4;
  int bq = (int)(base >> 12);            // /K_
  float m = ws_m[bq], li = ws_linv[bq];
  float4 v = *(float4*)(attn + base);
  v.x = expf(v.x - m) * li;
  v.y = expf(v.y - m) * li;
  v.z = expf(v.z - m) * li;
  v.w = expf(v.w - m) * li;
  *(float4*)(attn + base) = v;
}

// ---------------------------------------------------------------------------
// Kernel 3: out = tanh(concat(mix, output) @ W^T + b)
//   grid = (B_*Q_)/QT = 256 blocks, block = 256 threads (8 waves)
//   GEMM: M = global q row, N = h (512), K = c (1024)
// ---------------------------------------------------------------------------
__global__ void __launch_bounds__(256)
out_gemm_kernel(const __bf16* __restrict__ mixbf,  // [B*Q,512]
                const __bf16* __restrict__ qbf,    // [B*Q,512]
                const __bf16* __restrict__ wbf,    // [512,1024]
                const float* __restrict__ bias,    // [512]
                float* __restrict__ out)           // [B*Q,512]
{
  const int tid = threadIdx.x, lane = tid & 31, w = tid >> 5;
  const int mh = w >> 2, hq = w & 3;
  const int lhalf = lane >> 4, l15 = lane & 15;
  const int r0 = blockIdx.x * QT;
  const int arow = r0 + mh * 16 + l15;

  v8f acc[8];
  #pragma unroll
  for (int i = 0; i < 8; ++i) { v8f z = {}; acc[i] = z; }

  for (int cs = 0; cs < 32; ++cs) {      // K = 1024 in steps of 32
    int c0 = cs * 32 + lhalf * 16;
    const __bf16* ap = (cs < 16)
        ? (mixbf + (size_t)arow * H_ + c0)
        : (qbf   + (size_t)arow * H_ + (c0 - 512));
    v16bf a = ld16(ap, ap + 8);
    #pragma unroll
    for (int nt = 0; nt < 8; ++nt) {
      int h = hq * 128 + nt * 16 + l15;
      v16bf bm = *(const v16bf*)(wbf + (size_t)h * 1024 + cs * 32 + lhalf * 16);
      acc[nt] = wmma_bf16(a, bm, acc[nt]);
    }
  }
  #pragma unroll
  for (int nt = 0; nt < 8; ++nt) {
    int h = hq * 128 + nt * 16 + l15;
    float bb = bias[h];
    #pragma unroll
    for (int d = 0; d < 8; ++d) {
      int m = mh * 16 + lhalf * 8 + d;
      out[(size_t)(r0 + m) * H_ + h] = tanhf(acc[nt][d] + bb);
    }
  }
}

// ---------------------------------------------------------------------------
// Launch
// ---------------------------------------------------------------------------
extern "C" void kernel_launch(void* const* d_in, const int* in_sizes, int n_in,
                              void* d_out, int out_size, void* d_ws, size_t ws_size,
                              hipStream_t stream) {
  (void)in_sizes; (void)n_in; (void)out_size; (void)ws_size;
  const float*   output  = (const float*)d_in[0];
  const float*   context = (const float*)d_in[1];
  const uint8_t* mask    = (const uint8_t*)d_in[2];
  const float*   W_out   = (const float*)d_in[3];
  const float*   b_out   = (const float*)d_in[4];

  float* outp = (float*)d_out;                       // [B,Q,H]
  float* attn = outp + (size_t)B_ * Q_ * H_;         // [B,Q,K]

  char* ws = (char*)d_ws;
  __bf16* ctxbf  = (__bf16*)(ws + 0);                // 33,554,432 B  [B,K,H]
  __bf16* ctxTbf = (__bf16*)(ws + 33554432);         // 33,554,432 B  [B,H,K]
  __bf16* qbf    = (__bf16*)(ws + 67108864);         //  8,388,608 B
  __bf16* wbf    = (__bf16*)(ws + 75497472);         //  1,048,576 B
  __bf16* mixbf  = (__bf16*)(ws + 76546048);         //  8,388,608 B
  float*  ws_m   = (float*)(ws + 84934656);          //     32,768 B
  float*  ws_li  = (float*)(ws + 84967424);          //     32,768 B

  // Kernel 0a: bf16 conversions
  cvt_bf16_kernel<<<(B_ * K_ * H_ / 4 + 255) / 256, 256, 0, stream>>>(
      context, ctxbf, B_ * K_ * H_ / 4);
  cvt_bf16_kernel<<<(B_ * Q_ * H_ / 4 + 255) / 256, 256, 0, stream>>>(
      output, qbf, B_ * Q_ * H_ / 4);
  cvt_bf16_kernel<<<(H_ * 2 * H_ / 4 + 255) / 256, 256, 0, stream>>>(
      W_out, wbf, H_ * 2 * H_ / 4);

  // Kernel 0b: one-time context transpose (bf16)
  transpose_ctx_kernel<<<dim3(K_ / 32, H_ / 32, B_), 256, 0, stream>>>(
      ctxbf, ctxTbf);

  // Kernel 1: fused scores + online softmax + mix
  attn_softmax_mix_kernel<<<B_ * (Q_ / QT), 256, SMEM_SZ, stream>>>(
      qbf, ctxbf, ctxTbf, mask, attn, mixbf, ws_m, ws_li);

  // Kernel 2: normalize attn in place
  softmax_norm_kernel<<<(int)(((size_t)B_ * Q_ * K_ / 4) / 256), 256, 0, stream>>>(
      attn, ws_m, ws_li);

  // Kernel 3: output GEMM + tanh
  out_gemm_kernel<<<(B_ * Q_) / QT, 256, 0, stream>>>(
      mixbf, qbf, wbf, b_out, outp);
}
